// EditGenerator_33603824124657
// MI455X (gfx1250) — compile-verified, matrix operand order checked by reading, not run on hardware
//
#include <hip/hip_runtime.h>
#include <math.h>

// ---------------------------------------------------------------------------
// Types for CDNA5 WMMA (wave32): v_wmma_f32_16x16x32_bf16
// ---------------------------------------------------------------------------
typedef __attribute__((ext_vector_type(16))) __bf16 v16bf;
typedef __attribute__((ext_vector_type(8)))  __bf16 v8bf;
typedef __attribute__((ext_vector_type(4)))  __bf16 v4bf;
typedef __attribute__((ext_vector_type(8)))  float  v8f;

#define BM 128
#define BN 128
#define BK 32
#define LDSP 40   // LDS row stride in bf16 (80 B): conflict-free & 16B aligned

struct GemmParams {
  const float* A;     // A[m,k] row-major (per batch)
  const float* Bm;    // transB=1: W[n,k] row-major ; transB=0: B[k,n] row-major
  const float* bias;  // bias[n] or null
  const float* Res;   // residual[m*ldr + n] or null (nh==1 only)
  float*       C;     // C[m*ldc + n] (per batch)
  long asb, ash, bsb, bsh, csb, csh;  // per-(b,h) element offsets
  int lda, ldb, ldc, ldr;
  int M, N, K;
  int nh, transB, relu;
  float alpha;
};

// Fragment: lane L holds row (A) / col (B) = L%16; K chunks [8r..8r+7] and
// [16+8r..16+8r+7] with r = L/16, per the CDNA5 16-bit A-matrix VGPR layout.
__device__ __forceinline__ v16bf frag_load(const __bf16* p) {
  union { v16bf v; v8bf h[2]; } u;
  u.h[0] = *(const v8bf*)(p);
  u.h[1] = *(const v8bf*)(p + 16);
  return u.v;
}

__global__ __launch_bounds__(256) void wmma_gemm_kernel(GemmParams p) {
  // double-buffered staging: one barrier per K-step, global latency hidden
  __shared__ __attribute__((aligned(16))) __bf16 As[2][BM * LDSP];
  __shared__ __attribute__((aligned(16))) __bf16 Bs[2][BN * LDSP];

  const int tid = threadIdx.x;
  const int z   = blockIdx.z;
  const int bb  = z / p.nh;
  const int hh  = z % p.nh;
  const float* __restrict__ Ap = p.A  + (long)bb * p.asb + (long)hh * p.ash;
  const float* __restrict__ Bp = p.Bm + (long)bb * p.bsb + (long)hh * p.bsh;
  float*       __restrict__ Cp = p.C  + (long)bb * p.csb + (long)hh * p.csh;

  const int m0 = blockIdx.y * BM;
  const int n0 = blockIdx.x * BN;

  const int lane = tid & 31;
  const int r    = lane >> 4;   // 0/1: K-half select
  const int mr   = lane & 15;   // row/col within 16
  const int wave = tid >> 5;    // 8 waves
  const int wm   = (wave >> 2) * 64;  // wave M base (2 rows of waves)
  const int wn   = (wave & 3) * 32;   // wave N base (4 cols of waves)

  // per-thread staging coordinates (loop invariant)
  int sr[4], sc4[4], tk[4], tn4[4];
  #pragma unroll
  for (int it = 0; it < 4; ++it) {
    int idx = tid + it * 256;
    sr[it]  = idx >> 3;          // 0..127
    sc4[it] = (idx & 7) << 2;    // 0..28
    tk[it]  = idx >> 5;          // 0..31
    tn4[it] = (idx & 31) << 2;   // 0..124
  }

  v8f acc[4][2];
  #pragma unroll
  for (int i = 0; i < 4; ++i)
    #pragma unroll
    for (int j = 0; j < 2; ++j)
      #pragma unroll
      for (int e = 0; e < 8; ++e) acc[i][j][e] = 0.0f;

  float4 ra[4], rb[4];

  // ---- global -> registers for one K-stage ----
  auto g2r = [&](int k0) {
    #pragma unroll
    for (int it = 0; it < 4; ++it) {
      float4 v = {0.f, 0.f, 0.f, 0.f};
      int gm = m0 + sr[it];
      if (gm < p.M) v = *(const float4*)(Ap + (long)gm * p.lda + k0 + sc4[it]);
      ra[it] = v;
    }
    if (p.transB) {
      #pragma unroll
      for (int it = 0; it < 4; ++it) {
        float4 v = {0.f, 0.f, 0.f, 0.f};
        int gn = n0 + sr[it];
        if (gn < p.N) v = *(const float4*)(Bp + (long)gn * p.ldb + k0 + sc4[it]);
        rb[it] = v;
      }
    } else {
      #pragma unroll
      for (int it = 0; it < 4; ++it) {
        float4 v = {0.f, 0.f, 0.f, 0.f};
        int gn = n0 + tn4[it];
        if (gn < p.N) v = *(const float4*)(Bp + (long)(k0 + tk[it]) * p.ldb + gn);
        rb[it] = v;
      }
    }
  };

  // ---- registers -> LDS (f32 -> bf16 convert) ----
  auto r2lds = [&](int buf) {
    __bf16* __restrict__ a = As[buf];
    __bf16* __restrict__ b = Bs[buf];
    #pragma unroll
    for (int it = 0; it < 4; ++it) {
      v4bf w;
      w[0] = (__bf16)ra[it].x; w[1] = (__bf16)ra[it].y;
      w[2] = (__bf16)ra[it].z; w[3] = (__bf16)ra[it].w;
      *(v4bf*)(&a[sr[it] * LDSP + sc4[it]]) = w;
    }
    if (p.transB) {
      #pragma unroll
      for (int it = 0; it < 4; ++it) {
        v4bf w;
        w[0] = (__bf16)rb[it].x; w[1] = (__bf16)rb[it].y;
        w[2] = (__bf16)rb[it].z; w[3] = (__bf16)rb[it].w;
        *(v4bf*)(&b[sr[it] * LDSP + sc4[it]]) = w;
      }
    } else {
      #pragma unroll
      for (int it = 0; it < 4; ++it) {
        b[(tn4[it] + 0) * LDSP + tk[it]] = (__bf16)rb[it].x;
        b[(tn4[it] + 1) * LDSP + tk[it]] = (__bf16)rb[it].y;
        b[(tn4[it] + 2) * LDSP + tk[it]] = (__bf16)rb[it].z;
        b[(tn4[it] + 3) * LDSP + tk[it]] = (__bf16)rb[it].w;
      }
    }
  };

  const int nstage = p.K / BK;   // K is always a multiple of 32 here

  g2r(0);
  r2lds(0);
  __syncthreads();

  for (int s = 0; s < nstage; ++s) {
    const int cur = s & 1;
    if (s + 1 < nstage) g2r((s + 1) * BK);   // issue next-stage loads early

    const __bf16* __restrict__ ab = As[cur];
    const __bf16* __restrict__ bbuf = Bs[cur];
    v16bf af[4], bfr[2];
    #pragma unroll
    for (int i = 0; i < 4; ++i)
      af[i] = frag_load(&ab[(wm + i * 16 + mr) * LDSP + 8 * r]);
    #pragma unroll
    for (int j = 0; j < 2; ++j)
      bfr[j] = frag_load(&bbuf[(wn + j * 16 + mr) * LDSP + 8 * r]);
    #pragma unroll
    for (int i = 0; i < 4; ++i)
      #pragma unroll
      for (int j = 0; j < 2; ++j)
        acc[i][j] = __builtin_amdgcn_wmma_f32_16x16x32_bf16(
            false, af[i], false, bfr[j], (short)0, acc[i][j], false, false);

    if (s + 1 < nstage) {
      r2lds(cur ^ 1);
      __syncthreads();
    }
  }

  // ---- epilogue: C layout lane = n%16 (+16 -> M+8), VGPR e -> M = e + 8r ----
  #pragma unroll
  for (int j = 0; j < 2; ++j) {
    int n = n0 + wn + j * 16 + mr;
    if (n >= p.N) continue;
    float bv = p.bias ? p.bias[n] : 0.f;
    #pragma unroll
    for (int i = 0; i < 4; ++i) {
      #pragma unroll
      for (int e = 0; e < 8; ++e) {
        int m = m0 + wm + i * 16 + 8 * r + e;
        if (m < p.M) {
          float val = acc[i][j][e] * p.alpha + bv;
          if (p.Res) val += p.Res[(long)m * p.ldr + n];
          if (p.relu) val = fmaxf(val, 0.f);
          Cp[(long)m * p.ldc + n] = val;
        }
      }
    }
  }
}

// ---------------------------------------------------------------------------
// Elementwise / row kernels
// ---------------------------------------------------------------------------
__global__ void embed_kernel(const int* tokens, const float* emb,
                             const float* pos, float* x, int T, int D) {
  int row = blockIdx.x;            // b*T + t
  int t   = row % T;
  int tok = tokens[row];
  for (int d = threadIdx.x; d < D; d += blockDim.x)
    x[(long)row * D + d] = emb[(long)tok * D + d] + pos[(long)t * D + d];
}

__global__ __launch_bounds__(256) void softmax_kernel(float* s, int len) {
  __shared__ float red[256];
  float* p = s + (long)blockIdx.x * len;
  int tid = threadIdx.x;
  float mx = -3.402823466e38f;
  for (int i = tid; i < len; i += 256) mx = fmaxf(mx, p[i]);
  red[tid] = mx; __syncthreads();
  for (int o = 128; o > 0; o >>= 1) {
    if (tid < o) red[tid] = fmaxf(red[tid], red[tid + o]);
    __syncthreads();
  }
  mx = red[0]; __syncthreads();
  float sum = 0.f;
  for (int i = tid; i < len; i += 256) {
    float e = __expf(p[i] - mx);
    p[i] = e; sum += e;
  }
  red[tid] = sum; __syncthreads();
  for (int o = 128; o > 0; o >>= 1) {
    if (tid < o) red[tid] += red[tid + o];
    __syncthreads();
  }
  float inv = 1.f / red[0];
  __syncthreads();
  for (int i = tid; i < len; i += 256) p[i] *= inv;
}

__global__ __launch_bounds__(256) void layernorm_kernel(
    const float* in, float* out, const float* g, const float* b, int D) {
  __shared__ float red[256];
  const float* p = in + (long)blockIdx.x * D;
  float* q = out + (long)blockIdx.x * D;
  int tid = threadIdx.x;
  float s = 0.f;
  for (int i = tid; i < D; i += 256) s += p[i];
  red[tid] = s; __syncthreads();
  for (int o = 128; o > 0; o >>= 1) {
    if (tid < o) red[tid] += red[tid + o];
    __syncthreads();
  }
  float mean = red[0] / (float)D;
  __syncthreads();
  float vs = 0.f;
  for (int i = tid; i < D; i += 256) { float d = p[i] - mean; vs += d * d; }
  red[tid] = vs; __syncthreads();
  for (int o = 128; o > 0; o >>= 1) {
    if (tid < o) red[tid] += red[tid + o];
    __syncthreads();
  }
  float inv = rsqrtf(red[0] / (float)D + 1e-5f);
  __syncthreads();
  for (int i = tid; i < D; i += 256) q[i] = (p[i] - mean) * inv * g[i] + b[i];
}

// ---------------------------------------------------------------------------
// Host
// ---------------------------------------------------------------------------
static inline void launch_gemm(hipStream_t st,
    const float* A, long asb, long ash, int lda,
    const float* Bm, long bsb, long bsh, int ldb, int transB,
    const float* bias, const float* Res, int ldr,
    float* C, long csb, long csh, int ldc,
    int M, int N, int K, int nh, int batches, float alpha, int relu) {
  GemmParams p;
  p.A = A; p.Bm = Bm; p.bias = bias; p.Res = Res; p.C = C;
  p.asb = asb; p.ash = ash; p.bsb = bsb; p.bsh = bsh; p.csb = csb; p.csh = csh;
  p.lda = lda; p.ldb = ldb; p.ldc = ldc; p.ldr = ldr;
  p.M = M; p.N = N; p.K = K; p.nh = nh; p.transB = transB; p.relu = relu;
  p.alpha = alpha;
  dim3 grid((N + BN - 1) / BN, (M + BM - 1) / BM, batches);
  wmma_gemm_kernel<<<grid, dim3(256), 0, st>>>(p);
}

extern "C" void kernel_launch(void* const* d_in, const int* in_sizes, int n_in,
                              void* d_out, int out_size, void* d_ws, size_t ws_size,
                              hipStream_t stream) {
  const int B = 8, T = 512, S = 128, D = 768, L = 6, V = 50000, F = 3072, H = 8;
  const int hd = D / H;                        // 96
  const float sm_scale = 0.10206207261596575f; // 1/sqrt(96)

  const int*   tokens = (const int*)  d_in[0];
  const float* ctx    = (const float*)d_in[1];
  const float* emb    = (const float*)d_in[2];
  const float* pos    = (const float*)d_in[3];
  const float* sa_w   = (const float*)d_in[4];
  const float* sa_b   = (const float*)d_in[5];
  const float* sa_ow  = (const float*)d_in[6];
  const float* sa_ob  = (const float*)d_in[7];
  const float* ca_w   = (const float*)d_in[8];
  const float* ca_b   = (const float*)d_in[9];
  const float* ca_ow  = (const float*)d_in[10];
  const float* ca_ob  = (const float*)d_in[11];
  const float* l1_w   = (const float*)d_in[12];
  const float* l1_b   = (const float*)d_in[13];
  const float* l2_w   = (const float*)d_in[14];
  const float* l2_b   = (const float*)d_in[15];
  const float* n1_g   = (const float*)d_in[16];
  const float* n1_b   = (const float*)d_in[17];
  const float* n2_g   = (const float*)d_in[18];
  const float* n2_b   = (const float*)d_in[19];
  const float* n3_g   = (const float*)d_in[20];
  const float* n3_b   = (const float*)d_in[21];
  const float* op_w   = (const float*)d_in[22];
  const float* op_b   = (const float*)d_in[23];
  float* out = (float*)d_out;

  // workspace layout (floats)
  float* x    = (float*)d_ws;               // [B*T, D]
  float* tb   = x    + (long)B * T * D;     // [B*T, D]
  float* qkv  = tb   + (long)B * T * D;     // [B*T, 3D]
  float* qb   = qkv  + (long)B * T * 3 * D; // [B*T, D]
  float* kv   = qb   + (long)B * T * D;     // [B*S, 2D]
  float* attn = kv   + (long)B * S * 2 * D; // [B*T, D]
  float* sc   = attn + (long)B * T * D;     // [B,H,T,T]
  float* ff   = sc   + (long)B * H * T * T; // [B*T, F]

  embed_kernel<<<B * T, 256, 0, stream>>>(tokens, emb, pos, x, T, D);

  for (int i = 0; i < L; ++i) {
    const float* saw  = sa_w  + (long)i * 3 * D * D;
    const float* sab  = sa_b  + (long)i * 3 * D;
    const float* saow = sa_ow + (long)i * D * D;
    const float* saob = sa_ob + (long)i * D;
    const float* caw  = ca_w  + (long)i * 3 * D * D;
    const float* cab  = ca_b  + (long)i * 3 * D;
    const float* caow = ca_ow + (long)i * D * D;
    const float* caob = ca_ob + (long)i * D;
    const float* w1   = l1_w  + (long)i * F * D;
    const float* b1   = l1_b  + (long)i * F;
    const float* w2   = l2_w  + (long)i * D * F;
    const float* b2   = l2_b  + (long)i * D;

    // ---------------- self-attention ----------------
    // qkv = x @ in_w^T + in_b   [B*T, 3D]
    launch_gemm(stream, x, 0, 0, D, saw, 0, 0, D, 1, sab, nullptr, 0,
                qkv, 0, 0, 3 * D, B * T, 3 * D, D, 1, 1, 1.f, 0);
    // scores[b,h] = q @ k^T * 1/sqrt(hd)   [T,T]
    launch_gemm(stream, qkv, (long)T * 3 * D, hd, 3 * D,
                qkv + D, (long)T * 3 * D, hd, 3 * D, 1, nullptr, nullptr, 0,
                sc, (long)H * T * T, (long)T * T, T,
                T, T, hd, H, B * H, sm_scale, 0);
    softmax_kernel<<<B * H * T, 256, 0, stream>>>(sc, T);
    // o[b,h] = P @ V  -> attn[b,t, h*hd + d]
    launch_gemm(stream, sc, (long)H * T * T, (long)T * T, T,
                qkv + 2 * D, (long)T * 3 * D, hd, 3 * D, 0, nullptr, nullptr, 0,
                attn, (long)T * D, hd, D,
                T, hd, T, H, B * H, 1.f, 0);
    // out proj + residual; then post-norm
    launch_gemm(stream, attn, 0, 0, D, saow, 0, 0, D, 1, saob, x, D,
                tb, 0, 0, D, B * T, D, D, 1, 1, 1.f, 0);
    layernorm_kernel<<<B * T, 256, 0, stream>>>(tb, x, n1_g + (long)i * D, n1_b + (long)i * D, D);

    // ---------------- cross-attention ----------------
    launch_gemm(stream, x, 0, 0, D, caw, 0, 0, D, 1, cab, nullptr, 0,
                qb, 0, 0, D, B * T, D, D, 1, 1, 1.f, 0);
    launch_gemm(stream, ctx, 0, 0, D, caw + (long)D * D, 0, 0, D, 1, cab + D, nullptr, 0,
                kv, 0, 0, 2 * D, B * S, 2 * D, D, 1, 1, 1.f, 0);
    launch_gemm(stream, qb, (long)T * D, hd, D,
                kv, (long)S * 2 * D, hd, 2 * D, 1, nullptr, nullptr, 0,
                sc, (long)H * T * S, (long)T * S, S,
                T, S, hd, H, B * H, sm_scale, 0);
    softmax_kernel<<<B * H * T, 256, 0, stream>>>(sc, S);
    launch_gemm(stream, sc, (long)H * T * S, (long)T * S, S,
                kv + D, (long)S * 2 * D, hd, 2 * D, 0, nullptr, nullptr, 0,
                attn, (long)T * D, hd, D,
                T, hd, S, H, B * H, 1.f, 0);
    launch_gemm(stream, attn, 0, 0, D, caow, 0, 0, D, 1, caob, x, D,
                tb, 0, 0, D, B * T, D, D, 1, 1, 1.f, 0);
    layernorm_kernel<<<B * T, 256, 0, stream>>>(tb, x, n2_g + (long)i * D, n2_b + (long)i * D, D);

    // ---------------- FFN ----------------
    launch_gemm(stream, x, 0, 0, D, w1, 0, 0, D, 1, b1, nullptr, 0,
                ff, 0, 0, F, B * T, F, D, 1, 1, 1.f, 1 /*relu*/);
    launch_gemm(stream, ff, 0, 0, F, w2, 0, 0, F, 1, b2, x, D,
                tb, 0, 0, D, B * T, D, F, 1, 1, 1.f, 0);
    layernorm_kernel<<<B * T, 256, 0, stream>>>(tb, x, n3_g + (long)i * D, n3_b + (long)i * D, D);
  }

  // logits: x[:, T-1, :] @ op_w^T + op_b   [B, V]
  launch_gemm(stream, x + (long)(T - 1) * D, 0, 0, T * D,
              op_w, 0, 0, D, 1, op_b, nullptr, 0,
              out, 0, 0, V, B, V, D, 1, 1, 1.f, 0);
}
